// Grok1Attention_18210661335576
// MI455X (gfx1250) — matrix operand order repaired and checked by reading, not compiled, not run on hardware
//
#include <hip/hip_runtime.h>

typedef __attribute__((ext_vector_type(16))) __bf16 v16bf;
typedef __attribute__((ext_vector_type(8)))  __bf16 v8bf;
typedef __attribute__((ext_vector_type(8)))  float  v8f;

#define HIDDEN       4096
#define NUM_HEADS    32
#define NUM_KV_HEADS 8
#define HEAD_DIM     128
#define T_SEQ        2048
#define QKV_OUT      ((NUM_HEADS + 2*NUM_KV_HEADS)*HEAD_DIM)   // 6144
#define Q_SIZE       (NUM_HEADS*HEAD_DIM)                      // 4096
#define KV_SIZE      (NUM_KV_HEADS*HEAD_DIM)                   // 1024
#define LOGIT_CAP    30.0f
#define SCALING      0.08838834764831845f                      // 128^-0.5
#define NEG_BIG      (-3.0e38f)

__device__ __forceinline__ v8f wmma_bf16(v16bf a, v16bf b, v8f c) {
  return __builtin_amdgcn_wmma_f32_16x16x32_bf16(
      false, a, false, b, (short)0, c, false, false);
}

// f32 -> bf16 via hardware convert (RNE on gfx1250)
__device__ __forceinline__ __bf16 f2bf(float f) { return (__bf16)f; }

// Async global->LDS 16B copy (per-lane), CDNA5 path.
// Builtin expects (v4i addrspace(1)*, v4i addrspace(3)*, imm offset, imm cpol).
// We build the AS pointers via integer casts: per ISA 10.2, a flat LDS address
// has the LDS byte offset in its low 32 bits, and AS1 is the plain 64-bit VA.
__device__ __forceinline__ void async_b128(const void* g, void* l) {
  typedef int v4i __attribute__((vector_size(16)));
  typedef __attribute__((address_space(1))) v4i* gptr_t;
  typedef __attribute__((address_space(3))) v4i* lptr_t;
#if __has_builtin(__builtin_amdgcn_global_load_async_to_lds_b128)
  __builtin_amdgcn_global_load_async_to_lds_b128(
      (gptr_t)(unsigned long long)g,
      (lptr_t)(unsigned)(unsigned long long)l, 0, 0);
#else
  asm volatile("global_load_async_to_lds_b128 %0, %1, off"
               :: "v"((unsigned)(unsigned long long)l), "v"(g) : "memory");
#endif
}

__device__ __forceinline__ void wait_async0() {
#if __has_builtin(__builtin_amdgcn_s_wait_asynccnt)
  __builtin_amdgcn_s_wait_asynccnt(0);
#else
  asm volatile("s_wait_asynccnt 0" ::: "memory");
#endif
}

// ---------------------------------------------------------------------------
// C[M,N] = A[M,K] * B[N,K]^T   (f32 in/out, bf16 WMMA compute)
// BM=64, BN=128, BK=32, 128 threads (4 waves).  grid = (N/128, M/64).
// ---------------------------------------------------------------------------
__global__ __launch_bounds__(128) void gemm_bt_kernel(
    const float* __restrict__ A, const float* __restrict__ B,
    float* __restrict__ C, int M, int N, int K)
{
  constexpr int BM = 64, BN = 128, BK = 32;
  __shared__ __bf16 As[BM][BK];
  __shared__ __bf16 Bs[BN][BK];

  const int tid   = threadIdx.x;
  const int lane  = tid & 31;
  const int wave  = tid >> 5;
  const int hlane = lane & 15;
  const int half  = lane >> 4;
  const int row0  = blockIdx.y * BM;
  const int col0  = blockIdx.x * BN;

  v8f acc[BN/16];
  #pragma unroll
  for (int i = 0; i < BN/16; i++) acc[i] = v8f{};

  for (int k0 = 0; k0 < K; k0 += BK) {
    // prefetch next K-step tiles into cache while we convert this one
    if (k0 + BK < K) {
      int idx = tid;
      __builtin_prefetch(A + (size_t)(row0 + (idx >> 1))*K + k0 + BK, 0, 1);
      __builtin_prefetch(B + (size_t)(col0 +  idx      )*K + k0 + BK, 0, 1);
    }
    // stage A tile (512 float4, 4 per thread) with hw cvt
    #pragma unroll
    for (int i = 0; i < 4; i++) {
      int idx = tid + i*128;
      int r = idx >> 3, kq = (idx & 7) << 2;
      float4 v = *(const float4*)(A + (size_t)(row0 + r)*K + k0 + kq);
      As[r][kq+0] = f2bf(v.x); As[r][kq+1] = f2bf(v.y);
      As[r][kq+2] = f2bf(v.z); As[r][kq+3] = f2bf(v.w);
    }
    // stage B tile (1024 float4, 8 per thread)
    #pragma unroll
    for (int i = 0; i < 8; i++) {
      int idx = tid + i*128;
      int r = idx >> 3, kq = (idx & 7) << 2;
      float4 v = *(const float4*)(B + (size_t)(col0 + r)*K + k0 + kq);
      Bs[r][kq+0] = f2bf(v.x); Bs[r][kq+1] = f2bf(v.y);
      Bs[r][kq+2] = f2bf(v.z); Bs[r][kq+3] = f2bf(v.w);
    }
    __syncthreads();

    // A fragment: row = hlane, K = half*16 + {0..15}
    v16bf a;
    {
      const __bf16* p0 = &As[wave*16 + hlane][half*16];
      #pragma unroll
      for (int i = 0; i < 8; i++) { a[i] = p0[i]; a[8+i] = p0[8+i]; }
    }
    // preload ALL B fragments, then issue WMMAs back-to-back
    v16bf bfr[BN/16];
    #pragma unroll
    for (int nt = 0; nt < BN/16; nt++)
      bfr[nt] = *(const v16bf*)&Bs[nt*16 + hlane][half*16];
    #pragma unroll
    for (int nt = 0; nt < BN/16; nt++)
      acc[nt] = wmma_bf16(a, bfr[nt], acc[nt]);
    __syncthreads();
  }

  #pragma unroll
  for (int nt = 0; nt < BN/16; nt++) {
    #pragma unroll
    for (int v = 0; v < 8; v++) {
      int r = row0 + wave*16 + half*8 + v;
      int c = col0 + nt*16 + hlane;
      C[(size_t)r*N + c] = acc[nt][v];
    }
  }
}

// ---------------------------------------------------------------------------
// NeoX RoPE: read f32 qkv, write bf16 qkv (q/k rotated, v converted). 1 blk/t.
// ---------------------------------------------------------------------------
__global__ __launch_bounds__(128) void rope_kernel(
    const int* __restrict__ pos, const float* __restrict__ qkv,
    __bf16* __restrict__ qkvb)
{
  const int t = blockIdx.x;
  const float p = (float)pos[t];
  const float* row = qkv + (size_t)t * QKV_OUT;
  __bf16* rowb = qkvb + (size_t)t * QKV_OUT;
  for (int it = threadIdx.x; it < 48*64; it += 128) {
    int hv  = it >> 6;                 // 0..39 = q+k (rotate), 40..47 = v
    int i   = it & 63;
    int off = hv * HEAD_DIM;
    float x1 = row[off + i];
    float x2 = row[off + 64 + i];
    if (hv < 40) {
      float inv_freq = __expf(-(float)i * (2.0f/HEAD_DIM) * 9.210340371976184f);
      float s, c;
      __sincosf(p * inv_freq, &s, &c);
      float r1 = x1*c - x2*s;
      float r2 = x2*c + x1*s;
      x1 = r1; x2 = r2;
    }
    rowb[off + i]      = f2bf(x1);
    rowb[off + 64 + i] = f2bf(x2);
  }
}

// ---------------------------------------------------------------------------
// Flash attention: grid (T/64, NUM_HEADS), 128 threads (4 waves).
// K tiles: async global->LDS copy (bf16, no conversion).  V: LDS transpose.
// ---------------------------------------------------------------------------
__global__ __launch_bounds__(128) void attn_kernel(
    const __bf16* __restrict__ qkvb, float* __restrict__ attn)
{
  constexpr int BQ = 64, BKV = 64;
  __shared__ __bf16 Kt[BKV][HEAD_DIM];   // [key][dim] 16 KB (async-filled)
  __shared__ __bf16 Vt[HEAD_DIM][BKV];   // [dim][key] 16 KB (transposed)
  __shared__ __bf16 Pl[4][16][BKV];      // per-wave P  8 KB

  const int tid   = threadIdx.x;
  const int lane  = tid & 31;
  const int wave  = tid >> 5;
  const int hlane = lane & 15;
  const int half  = lane >> 4;
  const int head  = blockIdx.y;
  const int q0    = blockIdx.x * BQ;
  const int qw    = q0 + wave*16;

  const int qoff = head * HEAD_DIM;
  const int koff = Q_SIZE + (head >> 2) * HEAD_DIM;
  const int voff = Q_SIZE + KV_SIZE + (head >> 2) * HEAD_DIM;

  // Q fragments straight from bf16 global: two b128 loads per chunk
  v16bf qf[4];
  {
    const __bf16* qrow = qkvb + (size_t)(qw + hlane) * QKV_OUT + qoff;
    #pragma unroll
    for (int c = 0; c < 4; c++) {
      int kb = c*32 + half*16;
      v8bf lo = *(const v8bf*)(qrow + kb);
      v8bf hi = *(const v8bf*)(qrow + kb + 8);
      #pragma unroll
      for (int i = 0; i < 8; i++) { qf[c][i] = lo[i]; qf[c][8+i] = hi[i]; }
    }
  }

  v8f o[8];
  #pragma unroll
  for (int i = 0; i < 8; i++) o[i] = v8f{};
  float m_run[8], l_run[8];
  #pragma unroll
  for (int v = 0; v < 8; v++) { m_run[v] = NEG_BIG; l_run[v] = 0.0f; }

  const int nkb = blockIdx.x + 1;
  for (int kb = 0; kb < nkb; kb++) {
    const __bf16* kg = qkvb + (size_t)(kb*BKV) * QKV_OUT;
    // K tile: 1024 x 16B async copies (8 per thread), no format conversion
    #pragma unroll
    for (int i = 0; i < 8; i++) {
      int idx = tid + i*128;
      int kk  = idx >> 4;
      int d8  = (idx & 15) * 8;
      async_b128(kg + (size_t)kk*QKV_OUT + koff + d8, &Kt[kk][d8]);
    }
    // V tile: bf16 load + LDS transpose scatter
    #pragma unroll
    for (int i = 0; i < 8; i++) {
      int idx = tid + i*128;
      int kk  = idx >> 4;
      int d8  = (idx & 15) * 8;
      v8bf v8 = *(const v8bf*)(kg + (size_t)kk*QKV_OUT + voff + d8);
      #pragma unroll
      for (int j = 0; j < 8; j++) Vt[d8+j][kk] = v8[j];
    }
    wait_async0();
    __syncthreads();

    if (kb*BKV <= qw + 15) {
      // S = Q K^T
      v8f s[4];
      #pragma unroll
      for (int sub = 0; sub < 4; sub++) {
        v8f a = v8f{};
        #pragma unroll
        for (int c = 0; c < 4; c++) {
          v16bf bfrag = *(const v16bf*)&Kt[sub*16 + hlane][c*32 + half*16];
          a = wmma_bf16(qf[c], bfrag, a);
        }
        s[sub] = a;
      }
      // scale + tanh cap + causal mask
      const int qrow = qw + half*8;
      #pragma unroll
      for (int sub = 0; sub < 4; sub++) {
        int kcol = kb*BKV + sub*16 + hlane;
        #pragma unroll
        for (int v = 0; v < 8; v++) {
          float x = LOGIT_CAP * tanhf(s[sub][v] * (SCALING / LOGIT_CAP));
          s[sub][v] = (kcol <= qrow + v) ? x : NEG_BIG;
        }
      }
      // online softmax (16-lane half reductions match C/D row layout)
      float alpha[8];
      #pragma unroll
      for (int v = 0; v < 8; v++) {
        float m = fmaxf(fmaxf(s[0][v], s[1][v]), fmaxf(s[2][v], s[3][v]));
        #pragma unroll
        for (int off = 1; off < 16; off <<= 1) m = fmaxf(m, __shfl_xor(m, off));
        float mn = fmaxf(m_run[v], m);
        alpha[v] = __expf(m_run[v] - mn);
        m_run[v] = mn;
        float ls = 0.0f;
        #pragma unroll
        for (int sub = 0; sub < 4; sub++) {
          float p = __expf(s[sub][v] - mn);
          s[sub][v] = p;
          ls += p;
        }
        #pragma unroll
        for (int off = 1; off < 16; off <<= 1) ls += __shfl_xor(ls, off);
        l_run[v] = l_run[v]*alpha[v] + ls;
      }
      #pragma unroll
      for (int v = 0; v < 8; v++)
        #pragma unroll
        for (int dt = 0; dt < 8; dt++) o[dt][v] *= alpha[v];

      // P: C-layout -> A-layout via per-wave LDS scratch
      #pragma unroll
      for (int sub = 0; sub < 4; sub++)
        #pragma unroll
        for (int v = 0; v < 8; v++)
          Pl[wave][half*8 + v][sub*16 + hlane] = f2bf(s[sub][v]);

      // O += P V
      #pragma unroll
      for (int c = 0; c < 2; c++) {
        v16bf pa;
        const __bf16* p0 = &Pl[wave][hlane][c*32 + half*16];
        #pragma unroll
        for (int i = 0; i < 8; i++) { pa[i] = p0[i]; pa[8+i] = p0[8+i]; }
        #pragma unroll
        for (int dt = 0; dt < 8; dt++) {
          v16bf vb = *(const v16bf*)&Vt[dt*16 + hlane][c*32 + half*16];
          o[dt] = wmma_bf16(pa, vb, o[dt]);
        }
      }
    }
    __syncthreads();
  }

  #pragma unroll
  for (int v = 0; v < 8; v++) {
    float inv_l = 1.0f / l_run[v];
    int r = qw + half*8 + v;
    #pragma unroll
    for (int dt = 0; dt < 8; dt++)
      attn[(size_t)r * Q_SIZE + qoff + dt*16 + hlane] = o[dt][v] * inv_l;
  }
}

// ---------------------------------------------------------------------------
extern "C" void kernel_launch(void* const* d_in, const int* in_sizes, int n_in,
                              void* d_out, int out_size, void* d_ws, size_t ws_size,
                              hipStream_t stream) {
  const int*   positions = (const int*)d_in[0];
  const float* hidden    = (const float*)d_in[1];
  const float* w_qkv     = (const float*)d_in[2];
  const float* w_o       = (const float*)d_in[3];
  float* out   = (float*)d_out;
  float* qkv   = (float*)d_ws;                               // T x 6144 f32
  float* attn  = qkv + (size_t)T_SEQ * QKV_OUT;              // T x 4096 f32
  __bf16* qkvb = (__bf16*)(attn + (size_t)T_SEQ * Q_SIZE);   // T x 6144 bf16

  dim3 blk(128);
  gemm_bt_kernel<<<dim3(QKV_OUT/128, T_SEQ/64), blk, 0, stream>>>(
      hidden, w_qkv, qkv, T_SEQ, QKV_OUT, HIDDEN);
  rope_kernel<<<dim3(T_SEQ), blk, 0, stream>>>(positions, qkv, qkvb);
  attn_kernel<<<dim3(T_SEQ/64, NUM_HEADS), blk, 0, stream>>>(qkvb, attn);
  gemm_bt_kernel<<<dim3(HIDDEN/128, T_SEQ/64), blk, 0, stream>>>(
      attn, w_o, out, T_SEQ, HIDDEN, Q_SIZE);
}
